// AtomTransformer_80917183857333
// MI455X (gfx1250) — matrix operand order repaired and checked by reading, not compiled, not run on hardware
//
#include <hip/hip_runtime.h>

typedef __attribute__((ext_vector_type(16))) _Float16 v16h;
typedef __attribute__((ext_vector_type(8)))  _Float16 v8h;
typedef __attribute__((ext_vector_type(8)))  float    v8f;

#define NATOM 2048
#define CDIM  128
#define NHIDD 256
#define NTOK  512
#define NQ    32
#define NKEY  128

__device__ __forceinline__ float sigmoidf_(float x) { return 1.f / (1.f + __expf(-x)); }

// ---------------------------------------------------------------------------
// CDNA5 async global->LDS copy (ASYNCcnt-tracked, bypasses VGPRs).
// Per-lane: copies 16 bytes from the lane's global address to the lane's LDS
// byte address. Completion: s_wait_asynccnt 0 (per wave) + barrier (cross-wave).
// ---------------------------------------------------------------------------
__device__ __forceinline__ void async_g2l_b128(unsigned int lds_addr, const void* gaddr) {
  asm volatile("global_load_async_to_lds_b128 %0, %1, off"
               :: "v"(lds_addr), "v"(gaddr)
               : "memory");
}
__device__ __forceinline__ void wait_async0() {
  asm volatile("s_wait_asynccnt 0x0" ::: "memory");
}
__device__ __forceinline__ unsigned int lds_addr_of(const void* p) {
  return (unsigned int)(size_t)p;   // LDS aperture: low 32 bits = LDS byte offset
}

// ---------------------------------------------------------------------------
// WMMA fragment loaders (CDNA5 ISA 7.12.2 layouts), f16, 16x16x32.
// A tile in LDS is row-major [M][K]; lane L owns row M=L&15, halves 0..7 at
// K = k0 + (L>=16?8:0) + 0..7 and halves 8..15 at +16..23.
// B is staged as W^T row-major [N][K]; lane L owns col N=L&15, halves 0..7 at
// K = k0 + (L>=16?8:0) + 0..7 and halves 8..15 at +16..23 (same as A).
// ---------------------------------------------------------------------------
__device__ __forceinline__ v16h frag_a(const _Float16* base, int ld, int m0, int k0) {
  int lane = threadIdx.x & 31;
  const _Float16* p = base + (size_t)(m0 + (lane & 15)) * ld + k0 + ((lane & 16) ? 8 : 0);
  v8h lo = *(const v8h*)p;
  v8h hi = *(const v8h*)(p + 16);
  v16h r;
#pragma unroll
  for (int i = 0; i < 8; ++i) { r[i] = lo[i]; r[i + 8] = hi[i]; }
  return r;
}

__device__ __forceinline__ v16h frag_b(const _Float16* base, int ld, int n0, int k0) {
  int lane = threadIdx.x & 31;
  const _Float16* p = base + (size_t)(n0 + (lane & 15)) * ld + k0 + ((lane & 16) ? 8 : 0);
  v8h lo = *(const v8h*)p;
  v8h hi = *(const v8h*)(p + 16);
  v16h r;
#pragma unroll
  for (int i = 0; i < 8; ++i) { r[i] = lo[i]; r[i + 8] = hi[i]; }
  return r;
}

// ---------------------------------------------------------------------------
// Generic WMMA GEMM: out[M,N] = epilogue(A[M,K](f16) @ WT[N,K]^T(f16)).
// Workgroup = 128 threads (4 waves) -> 16 rows x 128 cols; grid (M/16, N/128).
// Operand tiles staged via async global->LDS copies.
// Epilogue modes:
//  0: (acc + bias[col]) * scale        1: sigmoid(acc + bias[col])
//  2: extra[idx] * acc                 3: extra2[idx] + extra[idx] * acc
//  4: silu(extra[idx]) * acc
// Writes f16 (outH) if non-null, else f32 (outF).
// ---------------------------------------------------------------------------
__global__ __launch_bounds__(128) void gemm_kernel(
    const _Float16* __restrict__ A, const _Float16* __restrict__ WT,
    const float* __restrict__ bias, const float* __restrict__ extra,
    const float* __restrict__ extra2, float* __restrict__ outF,
    _Float16* __restrict__ outH, int M, int N, int K, float scale, int mode)
{
  __shared__ alignas(16) _Float16 As[16 * 128];
  __shared__ alignas(16) _Float16 Ws[128 * 128];
  const int tid = threadIdx.x;
  const int wave = tid >> 5;
  const int lane = tid & 31;
  const int r0 = blockIdx.x * 16;
  const int n0 = blockIdx.y * 128;

  v8f acc0 = {};
  v8f acc1 = {};

  for (int kc = 0; kc < K; kc += 128) {
    __syncthreads();               // all waves done reading previous chunk
    for (int i = tid; i < 16 * 16; i += 128) {       // 16 rows x 128 halves
      int row = i >> 4, c8 = i & 15;
      async_g2l_b128(lds_addr_of(&As[row * 128 + c8 * 8]),
                     &A[(size_t)(r0 + row) * K + kc + c8 * 8]);
    }
    for (int i = tid; i < 128 * 16; i += 128) {      // 128 cols x 128 halves
      int row = i >> 4, c8 = i & 15;
      async_g2l_b128(lds_addr_of(&Ws[row * 128 + c8 * 8]),
                     &WT[(size_t)(n0 + row) * K + kc + c8 * 8]);
    }
    wait_async0();
    __syncthreads();
#pragma unroll
    for (int kk = 0; kk < 128; kk += 32) {
      v16h a  = frag_a(As, 128, 0, kk);
      v16h b0 = frag_b(Ws, 128, wave * 32, kk);
      v16h b1 = frag_b(Ws, 128, wave * 32 + 16, kk);
      acc0 = __builtin_amdgcn_wmma_f32_16x16x32_f16(false, a, false, b0, (short)0, acc0, false, false);
      acc1 = __builtin_amdgcn_wmma_f32_16x16x32_f16(false, a, false, b1, (short)0, acc1, false, false);
    }
  }

  const int roff = (lane & 16) ? 8 : 0;
  const int cbase = n0 + wave * 32 + (lane & 15);
#pragma unroll
  for (int j = 0; j < 8; ++j) {
    int row = r0 + j + roff;
#pragma unroll
    for (int t = 0; t < 2; ++t) {
      int col = cbase + t * 16;
      float acc = t ? acc1[j] : acc0[j];
      size_t idx = (size_t)row * N + col;
      float bv = bias ? bias[col] : 0.f;
      float r;
      switch (mode) {
        case 0:  r = (acc + bv) * scale; break;
        case 1:  r = sigmoidf_(acc + bv); break;
        case 2:  r = extra[idx] * acc; break;
        case 3:  r = extra2[idx] + extra[idx] * acc; break;
        default: { float x = extra[idx]; r = (x * sigmoidf_(x)) * acc; } break;
      }
      if (outH) outH[idx] = (_Float16)r; else outF[idx] = r;
    }
  }
}

// ---------------------------------------------------------------------------
// Row LayerNorm over 128 channels (one block per row, 128 threads).
// mode 0: out = LN(x)*gamma (f16)         -> s_n = LN(cl)*gs
// mode 1: out = g1*LN(x) + g2 (f16)       -> AdaLN combine (g1 pre-sigmoided)
// ---------------------------------------------------------------------------
__global__ __launch_bounds__(128) void ln_kernel(
    const float* __restrict__ x, const float* __restrict__ gamma,
    const float* __restrict__ g1, const float* __restrict__ g2,
    _Float16* __restrict__ out, int mode)
{
  __shared__ float red[128];
  const int row = blockIdx.x, t = threadIdx.x;
  const size_t idx = (size_t)row * CDIM + t;
  float v = x[idx];
  red[t] = v; __syncthreads();
#pragma unroll
  for (int s = 64; s > 0; s >>= 1) { if (t < s) red[t] += red[t + s]; __syncthreads(); }
  float mean = red[0] * (1.f / CDIM);
  __syncthreads();
  float d = v - mean;
  red[t] = d * d; __syncthreads();
#pragma unroll
  for (int s = 64; s > 0; s >>= 1) { if (t < s) red[t] += red[t + s]; __syncthreads(); }
  float an = d * rsqrtf(red[0] * (1.f / CDIM) + 1e-5f);
  float r = (mode == 0) ? an * gamma[t] : (g1[idx] * an + g2[idx]);
  out[idx] = (_Float16)r;
}

__global__ void cvt_half_kernel(const float* __restrict__ s, _Float16* __restrict__ d, int n) {
  int i = blockIdx.x * 256 + threadIdx.x;
  if (i < n) d[i] = (_Float16)s[i];
}

// W[K][N] f32 -> WT[N][K] f16
__global__ void transpose_kernel(const float* __restrict__ w, _Float16* __restrict__ wt,
                                 int K, int N) {
  int i = blockIdx.x * 256 + threadIdx.x;
  if (i < K * N) { int k = i / N, n = i % N; wt[(size_t)n * K + k] = (_Float16)w[i]; }
}

// mask_bias[l] = (sum_i a2t[l,i]*tmask[i] - 1) * 1e9
__global__ void mask_kernel(const float* __restrict__ a2t, const float* __restrict__ tm,
                            float* __restrict__ mb) {
  int i = blockIdx.x * 256 + threadIdx.x;
  if (i < NATOM) {
    float s = 0.f;
    for (int t = 0; t < NTOK; ++t) s += a2t[(size_t)i * NTOK + t] * tm[t];
    mb[i] = (s - 1.f) * 1e9f;
  }
}

// ---------------------------------------------------------------------------
// Pair bias only on the sparse 32x128 band:
// pbuf[h][n][j] = LN(plm[n, m(n,j)])*gz+bz  dot  wz[:,h]  + mask_bias[m],
// where m(n,j) = 32*(n/32) - 48 + j; out-of-range keys get -1e9.
// ---------------------------------------------------------------------------
__global__ __launch_bounds__(256) void pair_bias_kernel(
    const float* __restrict__ plm, const float* __restrict__ gz,
    const float* __restrict__ bz, const float* __restrict__ wz,
    const float* __restrict__ maskb, float* __restrict__ pbuf)
{
  int gid = blockIdx.x * 256 + threadIdx.x;       // over NATOM*NKEY
  int n = gid >> 7, j = gid & 127;
  int m = (n >> 5) * 32 - 48 + j;
  float ph[4];
  if (m >= 0 && m < NATOM) {
    const float* p = plm + ((size_t)n * NATOM + m) * 16;
    float xv[16], s = 0.f;
#pragma unroll
    for (int c = 0; c < 16; ++c) { xv[c] = p[c]; s += xv[c]; }
    float mean = s * (1.f / 16.f), s2 = 0.f;
#pragma unroll
    for (int c = 0; c < 16; ++c) { float d = xv[c] - mean; s2 += d * d; }
    float rs = rsqrtf(s2 * (1.f / 16.f) + 1e-5f);
    float mb = maskb[m];
#pragma unroll
    for (int h = 0; h < 4; ++h) ph[h] = mb;
#pragma unroll
    for (int c = 0; c < 16; ++c) {
      float z = (xv[c] - mean) * rs * gz[c] + bz[c];
#pragma unroll
      for (int h = 0; h < 4; ++h) ph[h] += z * wz[c * 4 + h];
    }
  } else {
#pragma unroll
    for (int h = 0; h < 4; ++h) ph[h] = -1e9f;
  }
#pragma unroll
  for (int h = 0; h < 4; ++h)
    pbuf[(((size_t)h * NATOM) + n) * NKEY + j] = ph[h];
}

// ---------------------------------------------------------------------------
// Fused block-sparse attention: one WG (4 waves) per 32-query block.
// Per head: WMMA logits (C operand = pair-bias tile), LDS softmax,
// WMMA attn@V against transposed V tile, gated store (sigmoid(a_ln@wg)).
// Q and pair-bias tiles staged via async global->LDS; K/V stagings are
// bounds-checked (zero-fill) so they use the manual path.
// ---------------------------------------------------------------------------
__global__ __launch_bounds__(128) void attn_kernel(
    const _Float16* __restrict__ q, const _Float16* __restrict__ k,
    const _Float16* __restrict__ v, const float* __restrict__ pbuf,
    const float* __restrict__ gsig, _Float16* __restrict__ o)
{
  __shared__ alignas(16) _Float16 s_q[NQ * 32];
  __shared__ alignas(16) _Float16 s_k[NKEY * 32];
  __shared__ alignas(16) _Float16 s_vt[32 * NKEY];
  __shared__ alignas(16) _Float16 s_at[NQ * NKEY];
  __shared__ alignas(16) float s_sc[NQ * NKEY];
  const int tid = threadIdx.x, wave = tid >> 5, lane = tid & 31;
  const int r0 = blockIdx.x * NQ;
  const int kstart = blockIdx.x * NQ - 48;

  for (int h = 0; h < 4; ++h) {
    __syncthreads();
    // async: Q tile (32x32 halves) and pair-bias scores tile (32x128 f32)
    for (int i = tid; i < NQ * 4; i += 128) {
      int row = i >> 2, c8 = i & 3;
      async_g2l_b128(lds_addr_of(&s_q[row * 32 + c8 * 8]),
                     &q[(size_t)(r0 + row) * CDIM + h * 32 + c8 * 8]);
    }
    for (int i = tid; i < NQ * NKEY / 4; i += 128) {
      int n = i >> 5, c4 = i & 31;
      async_g2l_b128(lds_addr_of(&s_sc[n * NKEY + c4 * 4]),
                     &pbuf[(((size_t)h * NATOM) + (r0 + n)) * NKEY + c4 * 4]);
    }
    // manual (zero-filled) K and V^T tiles
    for (int i = tid; i < NKEY * 4; i += 128) {
      int row = i >> 2, c8 = i & 3;
      int gk = kstart + row;
      v8h val = {};
      if (gk >= 0 && gk < NATOM) val = *(const v8h*)&k[(size_t)gk * CDIM + h * 32 + c8 * 8];
      *(v8h*)&s_k[row * 32 + c8 * 8] = val;
    }
    for (int i = tid; i < 32 * NKEY; i += 128) {     // V transposed: [d][key]
      int d = i >> 7, key = i & 127;
      int gk = kstart + key;
      s_vt[d * NKEY + key] = (gk >= 0 && gk < NATOM)
                                 ? v[(size_t)gk * CDIM + h * 32 + d] : (_Float16)0.f;
    }
    wait_async0();
    __syncthreads();

    // logits: 2 M-tiles x 8 N-tiles, 4 per wave; K=32 (per-head dim)
    for (int t = wave; t < 16; t += 4) {
      int mt = t >> 3, nt = t & 7;
      int roff = (lane & 16) ? 8 : 0;
      int cc = nt * 16 + (lane & 15);
      v8f c;
#pragma unroll
      for (int j = 0; j < 8; ++j) c[j] = s_sc[(mt * 16 + j + roff) * NKEY + cc];
      v16h a  = frag_a(s_q, 32, mt * 16, 0);
      v16h bm = frag_b(s_k, 32, nt * 16, 0);
      c = __builtin_amdgcn_wmma_f32_16x16x32_f16(false, a, false, bm, (short)0, c, false, false);
#pragma unroll
      for (int j = 0; j < 8; ++j) s_sc[(mt * 16 + j + roff) * NKEY + cc] = c[j];
    }
    __syncthreads();

    // softmax over 128 keys, one thread per query row
    if (tid < NQ) {
      float mx = -3.0e38f;
      for (int j = 0; j < NKEY; ++j) mx = fmaxf(mx, s_sc[tid * NKEY + j]);
      float s = 0.f;
      for (int j = 0; j < NKEY; ++j) { float e = __expf(s_sc[tid * NKEY + j] - mx); s_sc[tid * NKEY + j] = e; s += e; }
      float inv = 1.f / s;
      for (int j = 0; j < NKEY; ++j) s_at[tid * NKEY + j] = (_Float16)(s_sc[tid * NKEY + j] * inv);
    }
    __syncthreads();

    // O = attn @ V : 2x2 tiles, one per wave; K=128 keys
    {
      int mt = wave >> 1, nt = wave & 1;
      v8f acc = {};
#pragma unroll
      for (int kk = 0; kk < NKEY; kk += 32) {
        v16h a  = frag_a(s_at, NKEY, mt * 16, kk);
        v16h bm = frag_b(s_vt, NKEY, nt * 16, kk);
        acc = __builtin_amdgcn_wmma_f32_16x16x32_f16(false, a, false, bm, (short)0, acc, false, false);
      }
      int roff = (lane & 16) ? 8 : 0;
      int col = h * 32 + nt * 16 + (lane & 15);
#pragma unroll
      for (int j = 0; j < 8; ++j) {
        int row = r0 + mt * 16 + j + roff;
        size_t idx = (size_t)row * CDIM + col;
        o[idx] = (_Float16)(gsig[idx] * acc[j]);
      }
    }
  }
}

// ---------------------------------------------------------------------------
// Host orchestration
// ---------------------------------------------------------------------------
extern "C" void kernel_launch(void* const* d_in, const int* in_sizes, int n_in,
                              void* d_out, int out_size, void* d_ws, size_t ws_size,
                              hipStream_t stream)
{
  (void)in_sizes; (void)n_in; (void)out_size; (void)ws_size;

  const float* ql      = (const float*)d_in[0];
  const float* cl      = (const float*)d_in[1];
  const float* plm     = (const float*)d_in[2];
  const float* a2t     = (const float*)d_in[3];
  const float* tmask   = (const float*)d_in[4];
  const float* ada1_gs = (const float*)d_in[5];
  const float* ada1_bs = (const float*)d_in[7];
  const float* bq      = (const float*)d_in[10];
  const float* gz      = (const float*)d_in[13];
  const float* bz      = (const float*)d_in[14];
  const float* wz      = (const float*)d_in[15];
  const float* bog1    = (const float*)d_in[19];
  const float* ada2_gs = (const float*)d_in[20];
  const float* ada2_bs = (const float*)d_in[22];
  const float* bog2    = (const float*)d_in[28];

  char* wsbase = (char*)d_ws;
  size_t off = 0;
  auto alloc = [&](size_t bytes) -> void* {
    void* p = wsbase + off;
    off += (bytes + 255) & ~(size_t)255;
    return p;
  };

  float*    a_cur = (float*)alloc((size_t)NATOM * CDIM * 4);
  _Float16* clh   = (_Float16*)alloc((size_t)NATOM * CDIM * 2);
  _Float16* snh   = (_Float16*)alloc((size_t)NATOM * CDIM * 2);
  _Float16* alnh  = (_Float16*)alloc((size_t)NATOM * CDIM * 2);
  _Float16* a2h   = (_Float16*)alloc((size_t)NATOM * CDIM * 2);
  _Float16* qh    = (_Float16*)alloc((size_t)NATOM * CDIM * 2);
  _Float16* kh    = (_Float16*)alloc((size_t)NATOM * CDIM * 2);
  _Float16* vh    = (_Float16*)alloc((size_t)NATOM * CDIM * 2);
  _Float16* oh    = (_Float16*)alloc((size_t)NATOM * CDIM * 2);
  _Float16* hidh  = (_Float16*)alloc((size_t)NATOM * NHIDD * 2);
  float* G1     = (float*)alloc((size_t)NATOM * CDIM * 4);
  float* G2     = (float*)alloc((size_t)NATOM * CDIM * 4);
  float* gsig   = (float*)alloc((size_t)NATOM * CDIM * 4);
  float* gate1  = (float*)alloc((size_t)NATOM * CDIM * 4);
  float* gate2  = (float*)alloc((size_t)NATOM * CDIM * 4);
  float* attout = (float*)alloc((size_t)NATOM * CDIM * 4);
  float* h1     = (float*)alloc((size_t)NATOM * NHIDD * 4);
  float* pbuf   = (float*)alloc((size_t)4 * NATOM * NKEY * 4);
  float* maskb  = (float*)alloc((size_t)NATOM * 4);

  // Weight slots: {ws, wsb, wq, wk, wv, wg, wo, wog1, a2ws, a2wsb, wog2, wt1, wt2, wt3}
  const int slot_in[14] = {6, 8, 9, 11, 12, 16, 17, 18, 21, 23, 27, 24, 25, 26};
  const int slot_k[14]  = {128,128,128,128,128,128,128,128,128,128,128,128,128,256};
  const int slot_n[14]  = {128,128,128,128,128,128,128,128,128,128,128,256,256,128};
  _Float16* WT[3][14];
  for (int b = 0; b < 3; ++b)
    for (int s = 0; s < 14; ++s) {
      int K = slot_k[s], N = slot_n[s];
      WT[b][s] = (_Float16*)alloc((size_t)K * N * 2);
      const float* src = (const float*)d_in[slot_in[s]] + (size_t)b * K * N;
      transpose_kernel<<<(K * N + 255) / 256, 256, 0, stream>>>(src, WT[b][s], K, N);
    }

  hipMemcpyAsync(a_cur, ql, (size_t)NATOM * CDIM * 4, hipMemcpyDeviceToDevice, stream);
  cvt_half_kernel<<<(NATOM * CDIM + 255) / 256, 256, 0, stream>>>(cl, clh, NATOM * CDIM);
  mask_kernel<<<(NATOM + 255) / 256, 256, 0, stream>>>(a2t, tmask, maskb);

  auto GEMM = [&](const _Float16* A, const _Float16* W, const float* bias,
                  const float* ex, const float* ex2, float* oF, _Float16* oH,
                  int M, int N, int K, float scale, int mode) {
    dim3 g(M / 16, N / 128);
    gemm_kernel<<<g, 128, 0, stream>>>(A, W, bias, ex, ex2, oF, oH, M, N, K, scale, mode);
  };

  const float invs = 0.17677669529663687f;   // 1/sqrt(CH=32)

  for (int b = 0; b < 3; ++b) {
    // pair bias on the sparse band (includes key mask and OOB -inf)
    pair_bias_kernel<<<(NATOM * NKEY) / 256, 256, 0, stream>>>(
        plm, gz + b * 16, bz + b * 16, wz + b * 64, maskb, pbuf);

    // --- AdaLN 1: a_ln = sigmoid(s_n@ws + bs) * LN(a) + s_n@wsb ---
    ln_kernel<<<NATOM, 128, 0, stream>>>(cl, ada1_gs + b * CDIM, nullptr, nullptr, snh, 0);
    GEMM(snh, WT[b][0], ada1_bs + b * CDIM, nullptr, nullptr, G1, nullptr, NATOM, CDIM, CDIM, 1.f, 1);
    GEMM(snh, WT[b][1], nullptr, nullptr, nullptr, G2, nullptr, NATOM, CDIM, CDIM, 1.f, 0);
    ln_kernel<<<NATOM, 128, 0, stream>>>(a_cur, nullptr, G1, G2, alnh, 1);

    // output gates: sigmoid(cl@wog + bog)
    GEMM(clh, WT[b][7],  bog1 + b * CDIM, nullptr, nullptr, gate1, nullptr, NATOM, CDIM, CDIM, 1.f, 1);
    GEMM(clh, WT[b][10], bog2 + b * CDIM, nullptr, nullptr, gate2, nullptr, NATOM, CDIM, CDIM, 1.f, 1);

    // Q (bias + 1/sqrt(ch) folded), K, V (f16), gate G = sigmoid(a_ln@wg)
    GEMM(alnh, WT[b][2], bq + b * CDIM, nullptr, nullptr, nullptr, qh, NATOM, CDIM, CDIM, invs, 0);
    GEMM(alnh, WT[b][3], nullptr, nullptr, nullptr, nullptr, kh, NATOM, CDIM, CDIM, 1.f, 0);
    GEMM(alnh, WT[b][4], nullptr, nullptr, nullptr, nullptr, vh, NATOM, CDIM, CDIM, 1.f, 0);
    GEMM(alnh, WT[b][5], nullptr, nullptr, nullptr, gsig, nullptr, NATOM, CDIM, CDIM, 1.f, 1);

    attn_kernel<<<NATOM / NQ, 128, 0, stream>>>(qh, kh, vh, pbuf, gsig, oh);

    // att_out = gate1 * (o @ wo)
    GEMM(oh, WT[b][6], nullptr, gate1, nullptr, attout, nullptr, NATOM, CDIM, CDIM, 1.f, 2);

    // --- AdaLN 2 ---
    ln_kernel<<<NATOM, 128, 0, stream>>>(cl, ada2_gs + b * CDIM, nullptr, nullptr, snh, 0);
    GEMM(snh, WT[b][8], ada2_bs + b * CDIM, nullptr, nullptr, G1, nullptr, NATOM, CDIM, CDIM, 1.f, 1);
    GEMM(snh, WT[b][9], nullptr, nullptr, nullptr, G2, nullptr, NATOM, CDIM, CDIM, 1.f, 0);
    ln_kernel<<<NATOM, 128, 0, stream>>>(a_cur, nullptr, G1, G2, a2h, 1);

    // --- SwiGLU transition + residual update ---
    GEMM(a2h, WT[b][11], nullptr, nullptr, nullptr, h1, nullptr, NATOM, NHIDD, CDIM, 1.f, 0);
    GEMM(a2h, WT[b][12], nullptr, h1, nullptr, nullptr, hidh, NATOM, NHIDD, CDIM, 1.f, 4);
    float* outp = (b == 2) ? (float*)d_out : a_cur;   // a = att_out + gate2*(hidden@wt3)
    GEMM(hidh, WT[b][13], nullptr, gate2, attout, outp, nullptr, NATOM, CDIM, NHIDD, 1.f, 3);
  }
}
